// GINModel_2190433321023
// MI455X (gfx1250) — matrix operand order repaired and checked by reading, not compiled, not run on hardware
//
#include <hip/hip_runtime.h>

// ---------------------------------------------------------------------------
// GIN forward for MI455X (gfx1250, wave32).
// GEMMs: v_wmma_f32_16x16x32_bf16, bf16 operands pre-converted in global mem.
//   - A fragments loaded straight from global (rows are wave-exclusive)
//   - 64 x K weight panel staged once per block in LDS via async-to-LDS,
//     then a barrier-free K loop.
//   - 2x register blocking in M: each B fragment from LDS feeds 2 WMMAs.
// Scatter/pool: fp32 global atomics. BN: hierarchical column reduction.
// ---------------------------------------------------------------------------

#define N_NODES    20000
#define N_EDGES    320000
#define NUM_GRAPHS 256
#define BN_EPS     1e-5f

typedef __attribute__((ext_vector_type(16))) __bf16 v16bf;
typedef __attribute__((ext_vector_type(8)))  __bf16 v8bf;
typedef __attribute__((ext_vector_type(8)))  float  v8f;
typedef int v4i __attribute__((vector_size(16)));   // matches builtin param type

#if defined(__has_builtin)
#if __has_builtin(__builtin_amdgcn_global_load_async_to_lds_b128)
#define HAVE_ASYNC_LDS 1
#endif
#endif

__device__ __forceinline__ unsigned short bf16bits(float f) {
  unsigned u = __builtin_bit_cast(unsigned, f);
  u += 0x7FFFu + ((u >> 16) & 1u);          // round-to-nearest-even
  return (unsigned short)(u >> 16);
}
__device__ __forceinline__ __bf16 f2bf(float f) {
  unsigned short s = bf16bits(f);
  return __builtin_bit_cast(__bf16, s);
}

// ---------------- elementwise helpers ----------------

__global__ void zero_f32(float* __restrict__ p, int n) {
  int i = blockIdx.x * blockDim.x + threadIdx.x;
  if (i < n) p[i] = 0.0f;
}

__global__ void copy_f32v4(const float4* __restrict__ s, float4* __restrict__ d, int n4) {
  int i = blockIdx.x * blockDim.x + threadIdx.x;
  if (i < n4) d[i] = s[i];
}

struct us4 { unsigned short x, y, z, w; };

__global__ void cvt_f32_bf16(const float4* __restrict__ s, us4* __restrict__ d, int n4) {
  int i = blockIdx.x * blockDim.x + threadIdx.x;
  if (i >= n4) return;
  float4 v = s[i];
  us4 o;
  o.x = bf16bits(v.x); o.y = bf16bits(v.y);
  o.z = bf16bits(v.z); o.w = bf16bits(v.w);
  d[i] = o;
}

// ---------------- edge scatter: agg[dst] += x[src] ----------------
__global__ void scatter_add_edges(const float* __restrict__ x,
                                  const int* __restrict__ src,
                                  const int* __restrict__ dst,
                                  float* __restrict__ agg,
                                  int logPer, int dim) {
  long idx = (long)blockIdx.x * blockDim.x + threadIdx.x;
  long total = (long)N_EDGES << logPer;
  if (idx >= total) return;
  int e  = (int)(idx >> logPer);
  int c4 = ((int)idx & ((1 << logPer) - 1)) << 2;
  int s = src[e], d = dst[e];
  float4 v = *(const float4*)&x[(long)s * dim + c4];
  float* p = &agg[(long)d * dim + c4];
  atomicAdd(p + 0, v.x);
  atomicAdd(p + 1, v.y);
  atomicAdd(p + 2, v.z);
  atomicAdd(p + 3, v.w);
}

// ---------------- WMMA GEMM: C[N,O] = act(A[N,K] @ W[O,K]^T + b) ----------------
// A, W bf16 row-major; K = 1<<logK (128/256/512). 128 threads = 4 waves.
// Block tile: 128 rows x 64 cols; each wave owns 32 rows (2 A frags, 8 accums).
// Weight panel (64 x K bf16, <=64KB) staged once in LDS (async), K loop has no
// barriers; every LDS B fragment feeds 2 WMMAs.
// mode bit0 = relu, bit1 = store bf16 (else fp32).
__global__ __launch_bounds__(128)
void gemm_bf16(const __bf16* __restrict__ A, const __bf16* __restrict__ W,
               const float* __restrict__ bias, void* __restrict__ Cout,
               int N, int logK, int O, int mode) {
  extern __shared__ __align__(16) __bf16 ldsW[];   // [64][K]
  const int K    = 1 << logK;
  const int tid  = threadIdx.x;
  const int lane = tid & 31;
  const int wave = tid >> 5;
  const int row0 = blockIdx.y * 128;
  const int col0 = blockIdx.x * 64;

  // ---- stage 64 x K weight panel (16B chunks, uniform loop, no guards) ----
  {
    const int chunks = (64 << logK) >> 3;          // total 16B chunks
    const int cmask  = (K >> 3) - 1;
    for (int i = tid; i < chunks; i += 128) {
      int r  = i >> (logK - 3);
      int c8 = (i & cmask) << 3;
      const __bf16* gp = W + (((long)(col0 + r)) << logK) + c8;
      __bf16*       lp = &ldsW[(r << logK) + c8];
#if HAVE_ASYNC_LDS
      __builtin_amdgcn_global_load_async_to_lds_b128(
          (__attribute__((address_space(1))) v4i*)(void*)gp,
          (__attribute__((address_space(3))) v4i*)(void*)lp, 0, 0);
#else
      *(uint4*)lp = *(const uint4*)gp;
#endif
    }
#if HAVE_ASYNC_LDS
    asm volatile("s_wait_asynccnt 0x0" ::: "memory");
#endif
  }
  __syncthreads();

  // ---- two A row-fragments per wave; clamp rows (garbage rows unstored) ----
  int m0 = row0 + wave * 32 + (lane & 15);
  int m1 = m0 + 16;
  if (m0 > N - 1) m0 = N - 1;
  if (m1 > N - 1) m1 = N - 1;
  const __bf16* arow0 = A + ((long)m0 << logK);
  const __bf16* arow1 = A + ((long)m1 << logK);
  const int klo   = (lane >> 4) * 16;  // A: elems 0..15 -> K=klo..klo+15
  const int kBoff = (lane >> 4) * 16;  // B: 16 consecutive K at (lane>>4)*16
  const int nb    = lane & 15;

  v8f acc0[4] = {};
  v8f acc1[4] = {};
  for (int k0 = 0; k0 < K; k0 += 32) {
    union { v16bf v; v8bf h[2]; } a0, a1;
    a0.h[0] = *(const v8bf*)&arow0[k0 + klo];
    a0.h[1] = *(const v8bf*)&arow0[k0 + 8 + klo];
    a1.h[0] = *(const v8bf*)&arow1[k0 + klo];
    a1.h[1] = *(const v8bf*)&arow1[k0 + 8 + klo];
    #pragma unroll
    for (int s = 0; s < 4; ++s) {
      v16bf b = *(const v16bf*)&ldsW[((nb + s * 16) << logK) + k0 + kBoff];
      acc0[s] = __builtin_amdgcn_wmma_f32_16x16x32_bf16(
                    false, a0.v, false, b, (short)0, acc0[s], false, false);
      acc1[s] = __builtin_amdgcn_wmma_f32_16x16x32_bf16(
                    false, a1.v, false, b, (short)0, acc1[s], false, false);
    }
  }

  // ---- epilogue: D layout -> row = r + 8*(lane>>4), col = lane&15 (+16*s) ----
  const int colw = lane & 15;
  const int rofs = (lane >> 4) * 8;
  #pragma unroll
  for (int g = 0; g < 2; ++g) {
    const v8f* acc = g ? acc1 : acc0;
    #pragma unroll
    for (int s = 0; s < 4; ++s) {
      int col = col0 + s * 16 + colw;          // grid.x = O/64 exact -> col < O
      float bv = bias[col];
      #pragma unroll
      for (int r = 0; r < 8; ++r) {
        int row = row0 + wave * 32 + g * 16 + rofs + r;
        if (row < N) {
          float v = acc[s][r] + bv;
          if (mode & 1) v = fmaxf(v, 0.0f);
          if (mode & 2) ((__bf16*)Cout)[(long)row * O + col] = f2bf(v);
          else          ((float*)Cout)[(long)row * O + col] = v;
        }
      }
    }
  }
}

// ---------------- BatchNorm ----------------
__global__ void bn_stats(const float* __restrict__ h, float* __restrict__ sum,
                         float* __restrict__ sumsq, int N, int O, int rowsPerBlock) {
  int c = blockIdx.x * blockDim.x + threadIdx.x;
  if (c >= O) return;
  int r0 = blockIdx.y * rowsPerBlock;
  int r1 = r0 + rowsPerBlock; if (r1 > N) r1 = N;
  float s = 0.0f, q = 0.0f;
  for (int r = r0; r < r1; ++r) {
    float v = h[(long)r * O + c];
    s += v; q += v * v;
  }
  atomicAdd(&sum[c], s);
  atomicAdd(&sumsq[c], q);
}

__global__ void bn_apply_relu(float* __restrict__ h, const float* __restrict__ sum,
                              const float* __restrict__ sumsq,
                              const float* __restrict__ gamma,
                              const float* __restrict__ beta,
                              int total, int shiftO, float invN) {
  int i = blockIdx.x * blockDim.x + threadIdx.x;
  if (i >= total) return;
  int c = i & ((1 << shiftO) - 1);
  float mu  = sum[c] * invN;
  float var = sumsq[c] * invN - mu * mu;           // biased var (jnp.var ddof=0)
  float sc  = gamma[c] * rsqrtf(var + BN_EPS);
  float v   = (h[i] - mu) * sc + beta[c];
  h[i] = fmaxf(v, 0.0f);
}

// ---------------- global mean pool ----------------
__global__ void pool_count(const int* __restrict__ batch, float* __restrict__ counts) {
  int i = blockIdx.x * blockDim.x + threadIdx.x;
  if (i < N_NODES) atomicAdd(&counts[batch[i]], 1.0f);
}

__global__ void pool_sum(const float* __restrict__ x, const int* __restrict__ batch,
                         float* __restrict__ pooled) {
  int idx = blockIdx.x * blockDim.x + threadIdx.x;      // (node, col4), dim = 512
  if (idx >= N_NODES * 128) return;
  int n  = idx >> 7;
  int c4 = (idx & 127) << 2;
  int g  = batch[n];
  float4 v = *(const float4*)&x[(long)n * 512 + c4];
  float* p = &pooled[(long)g * 512 + c4];
  atomicAdd(p + 0, v.x);
  atomicAdd(p + 1, v.y);
  atomicAdd(p + 2, v.z);
  atomicAdd(p + 3, v.w);
}

__global__ void pool_div(float* __restrict__ pooled, const float* __restrict__ counts) {
  int i = blockIdx.x * blockDim.x + threadIdx.x;        // 256*512
  if (i >= NUM_GRAPHS * 512) return;
  int g = i >> 9;
  pooled[i] = pooled[i] / fmaxf(counts[g], 1.0f);
}

// ---------------------------------------------------------------------------

extern "C" void kernel_launch(void* const* d_in, const int* in_sizes, int n_in,
                              void* d_out, int out_size, void* d_ws, size_t ws_size,
                              hipStream_t stream) {
  const float* x     = (const float*)d_in[0];
  const int*   ei    = (const int*)d_in[1];
  const int*   batch = (const int*)d_in[2];
  const int*   src   = ei;
  const int*   dst   = ei + N_EDGES;
  const float* wp    = (const float*)d_in[21];
  const float* bp    = (const float*)d_in[22];

  // ---- workspace carve ----
  char* w = (char*)d_ws;
  float*  buf1   = (float*)w;  w += (size_t)N_NODES * 512 * 4;   // layer0/2 out (fp32)
  float*  buf2   = (float*)w;  w += (size_t)N_NODES * 256 * 4;   // layer1 out (fp32)
  float*  agg    = (float*)w;  w += (size_t)N_NODES * 256 * 4;   // scatter accum (fp32)
  float*  pooled = (float*)w;  w += (size_t)NUM_GRAPHS * 512 * 4;
  float*  counts = (float*)w;  w += (size_t)NUM_GRAPHS * 4;
  float*  bsum   = (float*)w;  w += 512 * 4;
  float*  bsq    = (float*)w;  w += 512 * 4;
  __bf16* aggbf  = (__bf16*)w; w += (size_t)N_NODES * 256 * 2;
  __bf16* h1bf   = (__bf16*)w; w += (size_t)N_NODES * 512 * 2;
  __bf16* w1bf   = (__bf16*)w; w += (size_t)512 * 512 * 2;
  __bf16* w2bf   = (__bf16*)w; w += (size_t)512 * 512 * 2;
  __bf16* wpbf   = (__bf16*)w; w += (size_t)64 * 512 * 2;
  __bf16* poolbf = (__bf16*)w; w += (size_t)NUM_GRAPHS * 512 * 2;

  const int dims[4] = {128, 128, 256, 512};
  const int logd[4] = {7, 7, 8, 9};
  const float* cur = x;

  for (int l = 0; l < 3; ++l) {
    const int K = dims[l], logK = logd[l];
    const int O = dims[l + 1], logO = logd[l + 1];
    const float* w1 = (const float*)d_in[3 + 6 * l];
    const float* b1 = (const float*)d_in[4 + 6 * l];
    const float* w2 = (const float*)d_in[5 + 6 * l];
    const float* b2 = (const float*)d_in[6 + 6 * l];
    const float* gm = (const float*)d_in[7 + 6 * l];
    const float* be = (const float*)d_in[8 + 6 * l];
    float* out = (l == 1) ? buf2 : buf1;

    // agg = x_cur ; agg[dst] += x[src]
    {
      int n4 = N_NODES * K / 4;
      copy_f32v4<<<(n4 + 255) / 256, 256, 0, stream>>>((const float4*)cur, (float4*)agg, n4);
      int logPer = logK - 2;
      long total = (long)N_EDGES << logPer;
      scatter_add_edges<<<(int)((total + 255) / 256), 256, 0, stream>>>(cur, src, dst, agg, logPer, K);
    }
    // bf16 conversions (activations + both weight matrices)
    {
      int n4 = N_NODES * K / 4;
      cvt_f32_bf16<<<(n4 + 255) / 256, 256, 0, stream>>>((const float4*)agg, (us4*)aggbf, n4);
      int nw1 = O * K / 4;
      cvt_f32_bf16<<<(nw1 + 255) / 256, 256, 0, stream>>>((const float4*)w1, (us4*)w1bf, nw1);
      int nw2 = O * O / 4;
      cvt_f32_bf16<<<(nw2 + 255) / 256, 256, 0, stream>>>((const float4*)w2, (us4*)w2bf, nw2);
    }
    // h1 = relu(agg @ w1^T + b1), stored bf16
    {
      dim3 grid(O / 64, (N_NODES + 127) / 128);
      gemm_bf16<<<grid, 128, (size_t)(128 << logK), stream>>>(
          aggbf, w1bf, b1, h1bf, N_NODES, logK, O, /*relu|bf16*/ 3);
    }
    // out = h1 @ w2^T + b2, stored fp32 for BN
    {
      dim3 grid(O / 64, (N_NODES + 127) / 128);
      gemm_bf16<<<grid, 128, (size_t)(128 << logO), stream>>>(
          h1bf, w2bf, b2, out, N_NODES, logO, O, /*plain fp32*/ 0);
    }
    // BatchNorm (training stats) + ReLU, in place
    {
      zero_f32<<<4, 256, 0, stream>>>(bsum, 1024);   // bsum+bsq contiguous
      const int rowsPerBlock = 250;
      dim3 grid(O / 128, (N_NODES + rowsPerBlock - 1) / rowsPerBlock);
      bn_stats<<<grid, 128, 0, stream>>>(out, bsum, bsq, N_NODES, O, rowsPerBlock);
      int total = N_NODES * O;
      bn_apply_relu<<<(total + 255) / 256, 256, 0, stream>>>(
          out, bsum, bsq, gm, be, total, logO, 1.0f / (float)N_NODES);
    }
    cur = out;
  }

  // global mean pool over graphs (cur = buf1, dim 512)
  {
    int nz = NUM_GRAPHS * 512 + NUM_GRAPHS;          // pooled + counts contiguous
    zero_f32<<<(nz + 255) / 256, 256, 0, stream>>>(pooled, nz);
    pool_count<<<(N_NODES + 255) / 256, 256, 0, stream>>>(batch, counts);
    int total = N_NODES * 128;
    pool_sum<<<(total + 255) / 256, 256, 0, stream>>>(cur, batch, pooled);
    pool_div<<<(NUM_GRAPHS * 512 + 255) / 256, 256, 0, stream>>>(pooled, counts);
  }

  // out = relu(pooled @ wp^T + bp)   [256x512 @ 512x64]
  {
    int n4 = NUM_GRAPHS * 512 / 4;
    cvt_f32_bf16<<<(n4 + 255) / 256, 256, 0, stream>>>((const float4*)pooled, (us4*)poolbf, n4);
    int nwp = 64 * 512 / 4;
    cvt_f32_bf16<<<(nwp + 255) / 256, 256, 0, stream>>>((const float4*)wp, (us4*)wpbf, nwp);
    dim3 grid(1, NUM_GRAPHS / 64);
    gemm_bf16<<<grid, 128, (size_t)(128 << 9), stream>>>(
        poolbf, wpbf, bp, (float*)d_out, NUM_GRAPHS, 9, 64, /*relu fp32*/ 1);
  }
}
